// MultiheadAttention_15968688406988
// MI455X (gfx1250) — matrix operand order repaired and checked by reading, not compiled, not run on hardware
//
#include <hip/hip_runtime.h>
#include <math.h>

typedef _Float16 v16h __attribute__((ext_vector_type(16)));
typedef _Float16 v8h  __attribute__((ext_vector_type(8)));
typedef float    v8f  __attribute__((ext_vector_type(8)));
typedef unsigned int u32x4 __attribute__((ext_vector_type(4)));
typedef int          i32x4 __attribute__((ext_vector_type(4)));
typedef int          i32x8 __attribute__((ext_vector_type(8)));

#define NB 8            // batch
#define NS 1024         // sequence
#define ND 1024         // model dim
#define NH 16           // heads
#define HD 64           // head dim
#define NM (NB*NS)      // 8192 flattened rows

// ---------------- phase 0a: X f32 -> f16 (8 elems / thread, b128 in+out) ---
__global__ void cvt_f32_to_f16(const float* __restrict__ src,
                               _Float16* __restrict__ dst, int n) {
  int i = (blockIdx.x * blockDim.x + threadIdx.x) * 8;
  if (i < n) {
    v8f x = *(const v8f*)(src + i);
    v8h y;
    for (int j = 0; j < 8; ++j) y[j] = (_Float16)x[j];
    *(v8h*)(dst + i) = y;
  }
}

// ---------------- phase 0b: W [K x N] f32 -> Wt [N x K] f16 ----------------
__global__ void cvt_transpose_w(const float* __restrict__ W,
                                _Float16* __restrict__ Wt) {
  int i = blockIdx.x * blockDim.x + threadIdx.x;   // over ND*ND
  int k = i / ND;
  int n = i % ND;
  Wt[(size_t)n * ND + k] = (_Float16)W[i];
}

// ---------------- phase 1: projections via WMMA, 32x64 strip per wave ------
__global__ void proj_gemm(const _Float16* __restrict__ Xh,
                          const _Float16* __restrict__ Wt,   // 4 matrices back-to-back
                          _Float16* __restrict__ Qh,
                          _Float16* __restrict__ Kh,
                          _Float16* __restrict__ Vt,
                          float* __restrict__ Rp) {
  const int wave = (blockIdx.x * blockDim.x + threadIdx.x) >> 5;
  const int lane = threadIdx.x & 31;
  const int lo16 = lane & 15;
  const bool hi  = lane >= 16;

  const int stripsPerMat = (NM / 32) * (ND / 64);  // 256 * 16 = 4096
  const int mat = wave / stripsPerMat;
  const int rem = wave % stripsPerMat;
  const int ms  = rem / (ND / 64);                 // 32-row strip index
  const int ns  = rem % (ND / 64);                 // 64-col strip index

  const _Float16* xr0 = Xh + (size_t)(ms * 32 + lo16) * ND;
  const _Float16* xr1 = xr0 + (size_t)16 * ND;
  const _Float16* wbase = Wt + (size_t)mat * ND * ND;

  v8f acc[2][4] = {};
  for (int kk = 0; kk < ND; kk += 32) {
    int ab = kk + (hi ? 16 : 0);
    v8h a0l = *(const v8h*)(xr0 + ab);
    v8h a0h = *(const v8h*)(xr0 + ab + 8);
    v8h a1l = *(const v8h*)(xr1 + ab);
    v8h a1h = *(const v8h*)(xr1 + ab + 8);
    v16h a0 = __builtin_shufflevector(a0l, a0h, 0,1,2,3,4,5,6,7,8,9,10,11,12,13,14,15);
    v16h a1 = __builtin_shufflevector(a1l, a1h, 0,1,2,3,4,5,6,7,8,9,10,11,12,13,14,15);
    for (int t = 0; t < 4; ++t) {
      const _Float16* wt = wbase + (size_t)(ns * 64 + t * 16 + lo16) * ND;
      v16h b = *(const v16h*)(wt + kk + (hi ? 16 : 0));
      acc[0][t] = __builtin_amdgcn_wmma_f32_16x16x32_f16(false, a0, false, b,
                                                         (short)0, acc[0][t], false, false);
      acc[1][t] = __builtin_amdgcn_wmma_f32_16x16x32_f16(false, a1, false, b,
                                                         (short)0, acc[1][t], false, false);
    }
  }

  const int roff = hi ? 8 : 0;
  for (int mi = 0; mi < 2; ++mi) {
    for (int t = 0; t < 4; ++t) {
      const int col = ns * 64 + t * 16 + lo16;
      if (mat <= 1) {
        _Float16* dst = (mat == 0) ? Qh : Kh;
        for (int r = 0; r < 8; ++r) {
          int row = ms * 32 + mi * 16 + r + roff;
          dst[(size_t)row * ND + col] = (_Float16)acc[mi][t][r];
        }
      } else if (mat == 2) {
        int h   = col / HD;
        int dim = col % HD;
        for (int r = 0; r < 8; ++r) {
          int row = ms * 32 + mi * 16 + r + roff;
          int bb  = row >> 10;
          int ss  = row & (NS - 1);
          Vt[(((size_t)bb * NH + h) * HD + dim) * NS + ss] = (_Float16)acc[mi][t][r];
        }
      } else {
        for (int r = 0; r < 8; ++r) {
          int row = ms * 32 + mi * 16 + r + roff;
          Rp[(size_t)row * ND + col] = acc[mi][t][r];
        }
      }
    }
  }
}

// ---- TDM: DMA a 32x64-f16 tile (row stride ND halves) into LDS -----------
// D# group0/group1 bit layout per CDNA5 ISA ch.8; 2-D tensor, data_size=2B.
__device__ __forceinline__ void tdm_load_k_tile(const _Float16* gaddr,
                                                unsigned ldsByteAddr) {
  unsigned long long ga = (unsigned long long)(size_t)gaddr;
  u32x4 g0;
  g0[0] = 1u;                                   // count=1, user descriptor
  g0[1] = ldsByteAddr;                          // lds_addr (bytes)
  g0[2] = (unsigned)(ga & 0xffffffffu);         // global_addr[31:0]
  g0[3] = (unsigned)((ga >> 32) & 0x1ffffffu)   // global_addr[56:32]
        | (2u << 30);                           // type=2 ("image")
  i32x8 g1;
  g1[0] = (1 << 16);                            // data_size=2B; mask=0; no flags
  g1[1] = (HD & 0xffff) << 16;                  // tensor_dim0[15:0] = 64
  g1[2] = ((NS & 0xffff) << 16);                // dim0 hi=0 | tensor_dim1 lo = 1024
  g1[3] = (HD << 16);                           // dim1 hi=0 | tile_dim0 = 64
  g1[4] = 32;                                   // tile_dim1 = 32 keys; tile_dim2=0
  g1[5] = ND;                                   // tensor_dim0_stride = 1024
  g1[6] = 0;                                    // stride0 hi | stride1 lo
  g1[7] = 0;
  i32x4 z4 = {0, 0, 0, 0};
#if defined(__clang_major__) && (__clang_major__ >= 23)
  i32x8 z8 = {0, 0, 0, 0, 0, 0, 0, 0};
  __builtin_amdgcn_tensor_load_to_lds(g0, g1, z4, z4, z8, 0);
#else
  __builtin_amdgcn_tensor_load_to_lds(g0, g1, z4, z4, 0);
#endif
}

// ---------------- phase 2: flash attention + residual + relu ----------------
// One block = 4 waves = 4 consecutive 16-query tiles of the same (b,h).
// Per 32-key step: K tile staged via TDM (wave 0), V tile via per-lane
// async global->LDS b128 loads; all 4 waves build WMMA B-fragments from LDS.
__global__ void attn_kernel(const _Float16* __restrict__ Qh,
                            const _Float16* __restrict__ Kh,
                            const _Float16* __restrict__ Vt,
                            const float* __restrict__ Rp,
                            float* __restrict__ out) {
  __shared__ __align__(32) _Float16 ldsK[32 * 64];      // [key][dim]   4 KB
  __shared__ __align__(32) _Float16 ldsV[64 * 32];      // [dim][key]   4 KB
  __shared__ __align__(32) _Float16 ldsP[4][16 * 32];   // per-wave P   4 KB

  const int tid  = threadIdx.x;
  const int wInB = tid >> 5;
  const int lane = tid & 31;
  const int lo16 = lane & 15;
  const bool hi  = lane >= 16;

  const int blk = blockIdx.x;
  const int b   = blk / (NH * (NS / 64));
  const int rem = blk % (NH * (NS / 64));
  const int h   = rem / (NS / 64);
  const int qg  = rem % (NS / 64);
  const int q0  = qg * 64 + wInB * 16;

  // Q tile 16x64 -> two A fragments (K=32 each); lane row q0+lo16
  const _Float16* qrow = Qh + ((size_t)(b * NS + q0 + lo16)) * ND + h * HD;
  v16h aQ[2];
  for (int f = 0; f < 2; ++f) {
    int ab = f * 32 + (hi ? 16 : 0);
    v8h a0 = *(const v8h*)(qrow + ab);
    v8h a1 = *(const v8h*)(qrow + ab + 8);
    aQ[f] = __builtin_shufflevector(a0, a1, 0,1,2,3,4,5,6,7,8,9,10,11,12,13,14,15);
  }

  float mrow[8], lrow[8];
  for (int r = 0; r < 8; ++r) { mrow[r] = -1e30f; lrow[r] = 0.0f; }
  v8f acc[4] = {};

  const float scale = 0.125f;            // 1/sqrt(HD)
  _Float16* myP = ldsP[wInB];
  const _Float16* vhead = Vt + ((size_t)b * NH + h) * HD * NS;

  for (int k0 = 0; k0 < NS; k0 += 32) {
    // ---- K tile via Tensor Data Mover (one wave issues; EXEC ignored) ----
    if (wInB == 0) {
      tdm_load_k_tile(Kh + (size_t)(b * NS + k0) * ND + h * HD,
                      (unsigned)(size_t)(void*)ldsK);
    }
    // ---- V tile via async global->LDS (2 x 16B chunks per thread) ----
    for (int c = tid; c < 256; c += 128) {
      int dim = c >> 2, vc8 = (c & 3) * 8;            // 8-half chunks
      unsigned la = (unsigned)(size_t)(void*)(ldsV + dim * 32 + vc8);
      const _Float16* gp = vhead + (size_t)dim * NS + k0 + vc8;
      asm volatile("global_load_async_to_lds_b128 %0, %1, off"
                   :: "v"(la), "v"(gp) : "memory");
    }
    asm volatile("s_wait_asynccnt 0" ::: "memory");    // own ASYNCcnt drained
    if (wInB == 0) __builtin_amdgcn_s_wait_tensorcnt(0);
    __syncthreads();                                   // publish K/V tiles

    // ---- scores: two 16x16 tiles covering keys k0..k0+31 (B from LDS) ----
    v8f sc[2] = {};
    for (int t = 0; t < 2; ++t) {
      const _Float16* kbase = ldsK + (t * 16 + lo16) * 64;
      for (int f = 0; f < 2; ++f) {
        v16h bK = *(const v16h*)(kbase + f * 32 + (hi ? 16 : 0));
        sc[t] = __builtin_amdgcn_wmma_f32_16x16x32_f16(false, aQ[f], false, bK,
                                                       (short)0, sc[t], false, false);
      }
    }

    // ---- online softmax (row-wise over 32 keys) ----
    for (int r = 0; r < 8; ++r) {
      float s0 = sc[0][r] * scale;
      float s1 = sc[1][r] * scale;
      float mx = fmaxf(s0, s1);
      for (int msk = 1; msk < 16; msk <<= 1)
        mx = fmaxf(mx, __shfl_xor(mx, msk, 32));
      float mnew  = fmaxf(mrow[r], mx);
      float alpha = __expf(mrow[r] - mnew);
      float p0 = __expf(s0 - mnew);
      float p1 = __expf(s1 - mnew);
      float ps = p0 + p1;
      for (int msk = 1; msk < 16; msk <<= 1)
        ps += __shfl_xor(ps, msk, 32);
      lrow[r] = lrow[r] * alpha + ps;
      mrow[r] = mnew;
      for (int t = 0; t < 4; ++t) acc[t][r] *= alpha;
      int row = r + (hi ? 8 : 0);
      myP[row * 32 + lo16]      = (_Float16)p0;
      myP[row * 32 + 16 + lo16] = (_Float16)p1;
    }

    // ---- reload P as a 16x32 A fragment (per-wave LDS ops are in-order) ----
    const _Float16* pr = myP + lo16 * 32;
    int ab = hi ? 16 : 0;
    v8h p0v = *(const v8h*)(pr + ab);
    v8h p1v = *(const v8h*)(pr + ab + 8);
    v16h aP = __builtin_shufflevector(p0v, p1v, 0,1,2,3,4,5,6,7,8,9,10,11,12,13,14,15);

    // ---- PV: 4 N-tiles of head dims (B from LDS) ----
    for (int t = 0; t < 4; ++t) {
      v16h bV = *(const v16h*)(ldsV + (t * 16 + lo16) * 32 + (hi ? 16 : 0));
      acc[t] = __builtin_amdgcn_wmma_f32_16x16x32_f16(false, aP, false, bV,
                                                      (short)0, acc[t], false, false);
    }
    __syncthreads();   // protect ldsK/ldsV before next iteration's staging
  }

  // ---- epilogue: O/l + residual projection, ReLU ----
  for (int t = 0; t < 4; ++t) {
    for (int r = 0; r < 8; ++r) {
      int q   = q0 + r + (hi ? 8 : 0);
      int col = h * HD + t * 16 + lo16;
      size_t idx = ((size_t)(b * NS + q)) * ND + col;
      float v = acc[t][r] / lrow[r] + Rp[idx];
      out[idx] = v > 0.0f ? v : 0.0f;
    }
  }
}

extern "C" void kernel_launch(void* const* d_in, const int* in_sizes, int n_in,
                              void* d_out, int out_size, void* d_ws, size_t ws_size,
                              hipStream_t stream) {
  const float* X  = (const float*)d_in[0];
  const float* WQ = (const float*)d_in[1];
  const float* WK = (const float*)d_in[2];
  const float* WV = (const float*)d_in[3];
  const float* WR = (const float*)d_in[4];
  float* out = (float*)d_out;

  char* ws = (char*)d_ws;
  size_t off = 0;
  _Float16* Xh = (_Float16*)(ws + off); off += (size_t)NM * ND * 2;       // 16 MB
  _Float16* Wt = (_Float16*)(ws + off); off += (size_t)4 * ND * ND * 2;   //  8 MB
  _Float16* Qh = (_Float16*)(ws + off); off += (size_t)NM * ND * 2;       // 16 MB
  _Float16* Kh = (_Float16*)(ws + off); off += (size_t)NM * ND * 2;       // 16 MB
  _Float16* Vt = (_Float16*)(ws + off); off += (size_t)NM * ND * 2;       // 16 MB
  float*    Rp = (float*)(ws + off);    off += (size_t)NM * ND * 4;       // 32 MB

  {
    int n = NM * ND;
    cvt_f32_to_f16<<<(n / 8 + 255) / 256, 256, 0, stream>>>(X, Xh, n);
    int nw = ND * ND;
    cvt_transpose_w<<<(nw + 255) / 256, 256, 0, stream>>>(WQ, Wt + 0 * (size_t)nw);
    cvt_transpose_w<<<(nw + 255) / 256, 256, 0, stream>>>(WK, Wt + 1 * (size_t)nw);
    cvt_transpose_w<<<(nw + 255) / 256, 256, 0, stream>>>(WV, Wt + 2 * (size_t)nw);
    cvt_transpose_w<<<(nw + 255) / 256, 256, 0, stream>>>(WR, Wt + 3 * (size_t)nw);
  }
  {
    int strips = 4 * (NM / 32) * (ND / 64);
    proj_gemm<<<strips / 8, 256, 0, stream>>>(Xh, Wt, Qh, Kh, Vt, Rp);
  }
  {
    int blocks = NB * NH * (NS / 64);
    attn_kernel<<<blocks, 128, 0, stream>>>(Qh, Kh, Vt, Rp, out);
  }
}
